// MultiHeadAttention_37022618092067
// MI455X (gfx1250) — matrix operand order repaired and checked by reading, not compiled, not run on hardware
//
#include <hip/hip_runtime.h>
#include <hip/hip_bf16.h>

#define B_  4
#define N_  256
#define ND  256   // NODE_DIM
#define ED  64    // EDGE_DIM
#define H_  8

typedef __attribute__((ext_vector_type(16))) __bf16 v16bf;
typedef __attribute__((ext_vector_type(8)))  __bf16 v8bf;
typedef __attribute__((ext_vector_type(8)))  float  v8f;

// D = A(16x32 bf16) * B(32x16 bf16) + C(16x16 f32)
__device__ __forceinline__ v8f wmma_bf16(v16bf a, v16bf b, v8f c) {
  return __builtin_amdgcn_wmma_f32_16x16x32_bf16(
      false, a, false, b, (short)0, c, false, false);
}

__device__ __forceinline__ v16bf joinbf(v8bf lo, v8bf hi) {
  return __builtin_shufflevector(lo, hi, 0, 1, 2, 3, 4, 5, 6, 7,
                                 8, 9, 10, 11, 12, 13, 14, 15);
}

// Build A operand from two contiguous 8-float runs (ISA 7.12.2 A layout:
// lane needs K = {hi..hi+7} and {16+hi..16+hi+7}, hi = 8*(lane>=16)).
__device__ __forceinline__ v16bf cvtA(v8f lo, v8f hi) {
  return joinbf(__builtin_convertvector(lo, v8bf),
                __builtin_convertvector(hi, v8bf));
}

// ---------------------------------------------------------------------------
// Pre-swizzle a fp32 [K x NC] matrix (nmat of them, contiguous) into bf16
// B-operand tiles: out[m][kt][ns][lane][i] = W[m][kt*32 + (lane>>4)*8 + (i&7) + (i>>3)*16][ns*16 + (lane&15)]
// Reader then does ONE aligned 32B load per B operand.
// ---------------------------------------------------------------------------
__global__ void __launch_bounds__(256)
k_swizzleB(const float* __restrict__ W, __bf16* __restrict__ out,
           int K, int NC, int nmat) {
  int per = K * NC;
  int total = nmat * per;
  int nsn = NC / 16;
  for (int idx = blockIdx.x * blockDim.x + threadIdx.x; idx < total;
       idx += gridDim.x * blockDim.x) {
    int m = idx / per;
    int r = idx - m * per;
    int i = r & 15;
    int lane = (r >> 4) & 31;
    int t = r >> 9;
    int ns = t % nsn;
    int kt = t / nsn;
    int krow = kt * 32 + (lane >> 4) * 8 + (i & 7) + (i >> 3) * 16;
    int col = ns * 16 + (lane & 15);
    out[idx] = (__bf16)W[(size_t)m * per + (size_t)krow * NC + col];
  }
}

// ---------------------------------------------------------------------------
// Y = (X[1024,256] @ W + bias) * scale ; Ws = pre-swizzled bf16 weight.
// 64 blocks x 8 waves; wave w owns cols [32w, 32w+32).
// ---------------------------------------------------------------------------
__global__ void __launch_bounds__(256)
k_proj(const float* __restrict__ X, const __bf16* __restrict__ Ws,
       const float* __restrict__ bias, float* __restrict__ Y, float scale) {
  int lane = threadIdx.x & 31;
  int w = threadIdx.x >> 5;
  int ln = lane & 15, g = lane >> 4, hi = g * 8;
  int row0 = blockIdx.x * 16;
  int col0 = w * 32 + ln, col1 = col0 + 16;
  const float* xrow = X + (size_t)(row0 + ln) * ND;
  v8f c0 = {}, c1 = {};
  for (int kt = 0; kt < 8; ++kt) {
    v16bf a = cvtA(*(const v8f*)(xrow + kt * 32 + hi),
                   *(const v8f*)(xrow + kt * 32 + 16 + hi));
    v16bf b0 = *(const v16bf*)(Ws + ((size_t)(kt * 16 + w * 2) * 32 + lane) * 16);
    v16bf b1 = *(const v16bf*)(Ws + ((size_t)(kt * 16 + w * 2 + 1) * 32 + lane) * 16);
    c0 = wmma_bf16(a, b0, c0);
    c1 = wmma_bf16(a, b1, c1);
  }
  float bv0 = bias[col0], bv1 = bias[col1];
  for (int r = 0; r < 8; ++r) {
    int orow = row0 + r + hi;
    Y[(size_t)orow * ND + col0] = (c0[r] + bv0) * scale;
    Y[(size_t)orow * ND + col1] = (c1[r] + bv1) * scale;
  }
}

// ---------------------------------------------------------------------------
// Fused edge pipeline. One block per (b,i); 8 waves; wave w = head h=w.
// edges read ONCE from HBM; new_edges never leaves the WGP.
// ---------------------------------------------------------------------------
__global__ void __launch_bounds__(256)
k_edge(const float* __restrict__ edges, const float* __restrict__ q,
       const float* __restrict__ kmat,
       const __bf16* __restrict__ Wm_s, const float* __restrict__ bm,
       const __bf16* __restrict__ Wa_s, const float* __restrict__ ba,
       const __bf16* __restrict__ We_s, const float* __restrict__ be,
       float* __restrict__ logits, float* __restrict__ out_edges) {
  __shared__ __align__(32) float  s_edge[32 * ED];   // 8 KB  fp32 edge tile
  __shared__ __align__(32) __bf16 s_ne[32 * ND];     // 16 KB leaky(new_edges)
  __shared__ __align__(32) __bf16 s_We[ND * ED];     // 32 KB swizzled We tiles

  int tid = threadIdx.x;
  int lane = tid & 31;
  int w = tid >> 5;  // head
  int ln = lane & 15, g = lane >> 4, hi = g * 8;

  int b = blockIdx.x >> 8;
  int i = blockIdx.x & 255;

  // stage pre-swizzled We -> LDS (pure uint4 copy, layout already B-operand)
  {
    const uint4* ps = (const uint4*)We_s;
    uint4* pd = (uint4*)s_We;
    for (int idx = tid; idx < (ND * ED) / 8; idx += 256) pd[idx] = ps[idx];
  }

  // resident B tiles for this head's 32 columns: one 32B load each
  v16bf Bm[2][2], Ba[2][2];
  for (int kt = 0; kt < 2; ++kt) {
    Bm[kt][0] = *(const v16bf*)(Wm_s + ((size_t)(kt * 16 + w * 2) * 32 + lane) * 16);
    Bm[kt][1] = *(const v16bf*)(Wm_s + ((size_t)(kt * 16 + w * 2 + 1) * 32 + lane) * 16);
    Ba[kt][0] = *(const v16bf*)(Wa_s + ((size_t)(kt * 16 + w * 2) * 32 + lane) * 16);
    Ba[kt][1] = *(const v16bf*)(Wa_s + ((size_t)(kt * 16 + w * 2 + 1) * 32 + lane) * 16);
  }

  int colA = w * 32 + ln;
  int colB = colA + 16;
  float qv0 = q[((size_t)b * N_ + i) * ND + colA];  // pre-scaled by 1/16
  float qv1 = q[((size_t)b * N_ + i) * ND + colB];
  float bm0 = bm[colA], bm1 = bm[colB];
  float ba0 = ba[colA], ba1 = ba[colB];

  const float* etile = edges + ((size_t)b * N_ + i) * N_ * ED;
  const float* krow = kmat + (size_t)b * N_ * ND;

  for (int it = 0; it < 8; ++it) {
    int jbase = it * 32;
    __syncthreads();
    {  // coalesced contiguous 32x64 fp32 tile load (8 KB)
      const float4* src = (const float4*)(etile + (size_t)jbase * ED);
      float4* dst = (float4*)s_edge;
      for (int idx = tid; idx < (32 * ED) / 4; idx += 256) dst[idx] = src[idx];
      if (it < 7) __builtin_prefetch(etile + (size_t)(jbase + 32) * ED, 0, 1);
    }
    __syncthreads();

    for (int rt = 0; rt < 2; ++rt) {
      v8f em0 = {}, em1 = {}, ea0 = {}, ea1 = {};
      const float* er = s_edge + (rt * 16 + ln) * ED;
      for (int kt = 0; kt < 2; ++kt) {
        v16bf a = cvtA(*(const v8f*)(er + kt * 32 + hi),
                       *(const v8f*)(er + kt * 32 + 16 + hi));
        em0 = wmma_bf16(a, Bm[kt][0], em0);
        em1 = wmma_bf16(a, Bm[kt][1], em1);
        ea0 = wmma_bf16(a, Ba[kt][0], ea0);
        ea1 = wmma_bf16(a, Ba[kt][1], ea1);
      }
      for (int r = 0; r < 8; ++r) {
        int jrow = jbase + rt * 16 + r + hi;
        float kv0 = krow[(size_t)jrow * ND + colA];
        float kv1 = krow[(size_t)jrow * ND + colB];
        float at0 = qv0 * kv0;
        float at1 = qv1 * kv1;
        float ne0 = at0 * (em0[r] + bm0) + at0 + (ea0[r] + ba0);
        float ne1 = at1 * (em1[r] + bm1) + at1 + (ea1[r] + ba1);
        // row-sum over this head's 32 cols (butterfly stays within 16-group)
        float sv = ne0 + ne1;
        sv += __shfl_xor(sv, 1, 32);
        sv += __shfl_xor(sv, 2, 32);
        sv += __shfl_xor(sv, 4, 32);
        sv += __shfl_xor(sv, 8, 32);
        if (ln == 0)
          logits[(((size_t)b * H_ + w) * N_ + i) * N_ + jrow] = sv;
        float l0 = ne0 > 0.f ? ne0 : 0.05f * ne0;
        float l1 = ne1 > 0.f ? ne1 : 0.05f * ne1;
        s_ne[(rt * 16 + r + hi) * ND + colA] = (__bf16)l0;
        s_ne[(rt * 16 + r + hi) * ND + colB] = (__bf16)l1;
      }
    }
    __syncthreads();

    // NE(32x256) @ We(256x64): wave -> (row tile w>>2, col subtile w&3)
    int rt2 = w >> 2;
    int cs = w & 3;
    int ocol = cs * 16 + ln;
    const __bf16* nr = s_ne + (rt2 * 16 + ln) * ND;
    v8f co = {};
    for (int kt = 0; kt < 8; ++kt) {
      v16bf a = joinbf(*(const v8bf*)(nr + kt * 32 + hi),
                       *(const v8bf*)(nr + kt * 32 + 16 + hi));
      v16bf bwe = *(const v16bf*)(s_We + ((size_t)(kt * 4 + cs) * 32 + lane) * 16);
      co = wmma_bf16(a, bwe, co);
    }
    float bev = be[ocol];
    for (int r = 0; r < 8; ++r) {
      int jrow = jbase + rt2 * 16 + r + hi;
      out_edges[(((size_t)b * N_ + i) * N_ + jrow) * ED + ocol] = co[r] + bev;
    }
  }
}

// ---------------------------------------------------------------------------
// Softmax over j: one wave per (b,h,i) row of 256 logits, in place.
// ---------------------------------------------------------------------------
__global__ void __launch_bounds__(256)
k_softmax(float* __restrict__ logits) {
  int lane = threadIdx.x & 31;
  int w = threadIdx.x >> 5;
  size_t row = (size_t)blockIdx.x * 8 + w;  // B*H*N = 8192 rows
  float* p = logits + row * N_;
  float v[8];
  float mx = -3.4e38f;
  for (int kk = 0; kk < 8; ++kk) {
    v[kk] = p[lane + kk * 32];
    mx = fmaxf(mx, v[kk]);
  }
  for (int m = 1; m < 32; m <<= 1) mx = fmaxf(mx, __shfl_xor(mx, m, 32));
  float sum = 0.f;
  for (int kk = 0; kk < 8; ++kk) {
    v[kk] = __expf(v[kk] - mx);
    sum += v[kk];
  }
  for (int m = 1; m < 32; m <<= 1) sum += __shfl_xor(sum, m, 32);
  float inv = 1.0f / sum;
  for (int kk = 0; kk < 8; ++kk) p[lane + kk * 32] = v[kk] * inv;
}

// ---------------------------------------------------------------------------
// weighted[b,i,h*32+d] = sum_j attn[b,h,i,j] * v[b,j,h*32+d]
// one wave per (b,h,i-tile); v pre-swizzled to B-operand tiles per batch.
// ---------------------------------------------------------------------------
__global__ void __launch_bounds__(32)
k_attnv(const float* __restrict__ attn, const __bf16* __restrict__ vs,
        float* __restrict__ weighted) {
  int lane = threadIdx.x & 31;
  int ln = lane & 15, g = lane >> 4, hi = g * 8;
  int it = blockIdx.x & 15;
  int h = (blockIdx.x >> 4) & 7;
  int b = blockIdx.x >> 7;
  const float* arow =
      attn + ((size_t)b * H_ + h) * N_ * N_ + (size_t)(it * 16 + ln) * N_;
  const __bf16* vb = vs + (size_t)b * ND * N_;
  int d0 = h * 32 + ln, d1 = d0 + 16;
  v8f c0 = {}, c1 = {};
  for (int kt = 0; kt < 8; ++kt) {
    v16bf a = cvtA(*(const v8f*)(arow + kt * 32 + hi),
                   *(const v8f*)(arow + kt * 32 + 16 + hi));
    v16bf b0 = *(const v16bf*)(vb + ((size_t)(kt * 16 + h * 2) * 32 + lane) * 16);
    v16bf b1 = *(const v16bf*)(vb + ((size_t)(kt * 16 + h * 2 + 1) * 32 + lane) * 16);
    c0 = wmma_bf16(a, b0, c0);
    c1 = wmma_bf16(a, b1, c1);
  }
  for (int r = 0; r < 8; ++r) {
    int ri = it * 16 + r + hi;
    weighted[((size_t)b * N_ + ri) * ND + d0] = c0[r];
    weighted[((size_t)b * N_ + ri) * ND + d1] = c1[r];
  }
}

// ---------------------------------------------------------------------------
extern "C" void kernel_launch(void* const* d_in, const int* in_sizes, int n_in,
                              void* d_out, int out_size, void* d_ws, size_t ws_size,
                              hipStream_t stream) {
  (void)in_sizes; (void)n_in; (void)out_size; (void)ws_size;
  const float* nodes = (const float*)d_in[0];
  const float* edges = (const float*)d_in[1];
  const float* Wq = (const float*)d_in[2];
  const float* bq = (const float*)d_in[3];
  const float* Wk = (const float*)d_in[4];
  const float* bk = (const float*)d_in[5];
  const float* Wv = (const float*)d_in[6];
  const float* bv = (const float*)d_in[7];
  const float* Wm = (const float*)d_in[8];
  const float* bm = (const float*)d_in[9];
  const float* Wa = (const float*)d_in[10];
  const float* ba = (const float*)d_in[11];
  const float* Wn = (const float*)d_in[12];
  const float* bn = (const float*)d_in[13];
  const float* We = (const float*)d_in[14];
  const float* be = (const float*)d_in[15];

  float* out_nodes = (float*)d_out;                     // [B,N,256]
  float* out_edges = out_nodes + (size_t)B_ * N_ * ND;  // [B,N,N,64]

  // fp32 workspace
  float* ws = (float*)d_ws;
  float* q = ws;                                     // 262144
  float* k = q + (size_t)B_ * N_ * ND;               // 262144
  float* v = k + (size_t)B_ * N_ * ND;               // 262144
  float* logits = v + (size_t)B_ * N_ * ND;          // 2097152
  float* weighted = logits + (size_t)B_ * H_ * N_ * N_;  // 262144
  // bf16 workspace (pre-swizzled B-operand tiles), 32B-aligned boundary
  __bf16* bws = (__bf16*)(weighted + (size_t)B_ * N_ * ND);
  __bf16* Wq_s = bws;                 // 65536
  __bf16* Wk_s = Wq_s + 65536;
  __bf16* Wv_s = Wk_s + 65536;
  __bf16* Wn_s = Wv_s + 65536;
  __bf16* Wm_s = Wn_s + 65536;        // 16384
  __bf16* Wa_s = Wm_s + 16384;        // 16384
  __bf16* We_s = Wa_s + 16384;        // 16384
  __bf16* v_s  = We_s + 16384;        // 262144

  const float scale = 1.0f / 16.0f;  // 1/sqrt(NODE_DIM)

  // weight pre-swizzles (tiny; L2-resident afterwards)
  k_swizzleB<<<64, 256, 0, stream>>>(Wq, Wq_s, 256, 256, 1);
  k_swizzleB<<<64, 256, 0, stream>>>(Wk, Wk_s, 256, 256, 1);
  k_swizzleB<<<64, 256, 0, stream>>>(Wv, Wv_s, 256, 256, 1);
  k_swizzleB<<<64, 256, 0, stream>>>(Wn, Wn_s, 256, 256, 1);
  k_swizzleB<<<64, 256, 0, stream>>>(Wm, Wm_s, 64, 256, 1);
  k_swizzleB<<<64, 256, 0, stream>>>(Wa, Wa_s, 64, 256, 1);
  k_swizzleB<<<64, 256, 0, stream>>>(We, We_s, 256, 64, 1);

  k_proj<<<64, 256, 0, stream>>>(nodes, Wq_s, bq, q, scale);
  k_proj<<<64, 256, 0, stream>>>(nodes, Wk_s, bk, k, 1.0f);
  k_proj<<<64, 256, 0, stream>>>(nodes, Wv_s, bv, v, 1.0f);
  k_swizzleB<<<256, 256, 0, stream>>>(v, v_s, 256, 256, 4);  // v -> B tiles

  k_edge<<<B_ * N_, 256, 0, stream>>>(edges, q, k, Wm_s, bm, Wa_s, ba, We_s, be,
                                      logits, out_edges);
  k_softmax<<<1024, 256, 0, stream>>>(logits);
  k_attnv<<<B_ * H_ * (N_ / 16), 32, 0, stream>>>(logits, v_s, weighted);
  k_proj<<<64, 256, 0, stream>>>(weighted, Wn_s, bn, out_nodes, 1.0f);
}
